// MambaBlock_51032801411306
// MI455X (gfx1250) — matrix operand order repaired and checked by reading, not compiled, not run on hardware
//
#include <hip/hip_runtime.h>
#include <math.h>

// ---------------------------------------------------------------------------
// Shapes (fixed by the reference)
// ---------------------------------------------------------------------------
#define HID   1024
#define INTER 2048
#define N2    4096          // 2*INTER
#define STATE 64
#define BATCH 4
#define SEQ   4096
#define TOK   (BATCH * SEQ) // 16384 tokens

// WMMA GEMM tiling
#define BM 128
#define BN 128
#define BK 64
#define AK 72               // padded LDS stride (halfs), keeps 16B row alignment

typedef __attribute__((ext_vector_type(16))) _Float16 v16h;
typedef __attribute__((ext_vector_type(8)))  float    v8f;
typedef __attribute__((ext_vector_type(4)))  unsigned uint4v;
typedef int gint4 __attribute__((vector_size(16)));   // matches async-LDS builtin param

#define AS1 __attribute__((address_space(1)))
#define AS3 __attribute__((address_space(3)))

// gfx1250 async global->LDS copy path (ASYNCcnt tracked), if the toolchain
// exposes the builtins. Falls back to VGPR round-trip staging otherwise.
#if defined(__has_builtin)
#if __has_builtin(__builtin_amdgcn_global_load_async_to_lds_b128) && \
    __has_builtin(__builtin_amdgcn_s_wait_asynccnt)
#define USE_ASYNC_COPY 1
#endif
#endif

__device__ __forceinline__ float silu_f(float v) { return v / (1.0f + __expf(-v)); }
__device__ __forceinline__ float softplus_f(float v) {
  return (v > 20.0f) ? v : log1pf(__expf(v));
}

// ---------------------------------------------------------------------------
// Small utility kernels
// ---------------------------------------------------------------------------
__global__ void k_f32_to_f16(const float* __restrict__ src, _Float16* __restrict__ dst, int n) {
  int i = blockIdx.x * 256 + threadIdx.x;
  if (i < n) dst[i] = (_Float16)src[i];
}

__global__ void k_addvec(const float* __restrict__ a, const float* __restrict__ b,
                         float* __restrict__ o, int n) {
  int i = blockIdx.x * 256 + threadIdx.x;
  if (i < n) o[i] = a[i] + b[i];
}

__global__ void k_zero(float* __restrict__ p, int n) {
  int i = blockIdx.x * 256 + threadIdx.x;
  if (i < n) p[i] = 0.0f;
}

// cond = condition @ W_cond + b_cond  (4 x 1024 x 2048, trivial)
__global__ void k_cond_gemm(const float* __restrict__ cnd, const float* __restrict__ W,
                            const float* __restrict__ bias, float* __restrict__ out) {
  int idx = blockIdx.x * 256 + threadIdx.x;      // 8192 outputs
  int b = idx >> 11;
  int j = idx & (INTER - 1);
  const float* cr = cnd + (size_t)b * HID;
  float acc = bias[j];
  for (int h = 0; h < HID; ++h) acc += cr[h] * W[(size_t)h * INTER + j];
  out[idx] = acc;
}

// ---------------------------------------------------------------------------
// LayerNorm per token -> f16 activation for GEMM1
// ---------------------------------------------------------------------------
__global__ __launch_bounds__(256) void k_layernorm(const float* __restrict__ x,
                                                   const float* __restrict__ g,
                                                   const float* __restrict__ bta,
                                                   _Float16* __restrict__ xn) {
  int t = blockIdx.x;
  const float* xr = x + (size_t)t * HID;
  __shared__ float red[256];
  float s = 0.f, s2 = 0.f;
  for (int i = threadIdx.x; i < HID; i += 256) { float v = xr[i]; s += v; s2 += v * v; }
  red[threadIdx.x] = s; __syncthreads();
  for (int o = 128; o > 0; o >>= 1) { if (threadIdx.x < o) red[threadIdx.x] += red[threadIdx.x + o]; __syncthreads(); }
  float mu = red[0] * (1.0f / HID); __syncthreads();
  red[threadIdx.x] = s2; __syncthreads();
  for (int o = 128; o > 0; o >>= 1) { if (threadIdx.x < o) red[threadIdx.x] += red[threadIdx.x + o]; __syncthreads(); }
  float var  = red[0] * (1.0f / HID) - mu * mu;
  float rstd = rsqrtf(var + 1e-5f);
  for (int i = threadIdx.x; i < HID; i += 256) {
    float v = (xr[i] - mu) * rstd * g[i] + bta[i];
    xn[(size_t)t * HID + i] = (_Float16)v;
  }
}

// ---------------------------------------------------------------------------
// Depthwise conv3 (pad 1 along L) + SiLU + cond add + per-token mean
// ---------------------------------------------------------------------------
__global__ __launch_bounds__(256) void k_conv_silu_cond(const float* __restrict__ xp0,
                                                        const float* __restrict__ cw,
                                                        const float* __restrict__ cb,
                                                        const float* __restrict__ cond,
                                                        float* __restrict__ xs,
                                                        _Float16* __restrict__ xsh,
                                                        float* __restrict__ xmean) {
  int t = blockIdx.x;
  int b = t / SEQ, l = t % SEQ;
  const float* cur = xp0 + (size_t)t * INTER;
  const float* prv = (l > 0)       ? cur - INTER : nullptr;
  const float* nxt = (l < SEQ - 1) ? cur + INTER : nullptr;
  const float* cnd = cond + (size_t)b * INTER;
  __shared__ float red[256];
  float s = 0.f;
  for (int c = threadIdx.x; c < INTER; c += 256) {
    float v = cw[c * 3 + 1] * cur[c] + cb[c];
    if (prv) v += cw[c * 3 + 0] * prv[c];
    if (nxt) v += cw[c * 3 + 2] * nxt[c];
    v = silu_f(v) + cnd[c];
    xs [(size_t)t * INTER + c] = v;
    xsh[(size_t)t * INTER + c] = (_Float16)v;
    s += v;
  }
  red[threadIdx.x] = s; __syncthreads();
  for (int o = 128; o > 0; o >>= 1) { if (threadIdx.x < o) red[threadIdx.x] += red[threadIdx.x + o]; __syncthreads(); }
  if (threadIdx.x == 0) xmean[t] = red[0] * (1.0f / INTER);
}

// ---------------------------------------------------------------------------
// Sequential selective scan: one block per batch, one lane per state
// ---------------------------------------------------------------------------
__global__ __launch_bounds__(64) void k_scan(const float* __restrict__ dtsum,
                                             const float* __restrict__ xmean,
                                             const float* __restrict__ Braw,
                                             const float* __restrict__ Cmat,
                                             const float* __restrict__ A_log,
                                             float* __restrict__ yscan) {
  int b = blockIdx.x, s = threadIdx.x;
  __shared__ float red[64];
  float An = -__expf(A_log[s]);
  float h = 0.f;
  for (int l = 0; l < SEQ; ++l) {
    int t = b * SEQ + l;
    float dt = dtsum[t] * (1.0f / INTER);   // mean of softplus row
    float a  = __expf(An * dt);
    float bx = Braw[(size_t)t * STATE + s] * dt * xmean[t];
    h = a * h + bx;
    red[s] = Cmat[(size_t)t * STATE + s] * h;
    __syncthreads();
    for (int o = 32; o > 0; o >>= 1) { if (s < o) red[s] += red[s + o]; __syncthreads(); }
    if (s == 0) yscan[t] = red[0];
    __syncthreads();
  }
}

// yh = (yscan + xs*Dp) * silu(gate)   -> f16 operand of output GEMM
__global__ void k_build_y(const float* __restrict__ yscan, const float* __restrict__ xs,
                          const float* __restrict__ Dp, const float* __restrict__ gsil,
                          _Float16* __restrict__ yh) {
  size_t i = (size_t)blockIdx.x * 256 + threadIdx.x;
  if (i >= (size_t)TOK * INTER) return;
  size_t t = i >> 11;
  int d = (int)(i & (INTER - 1));
  float v = (yscan[t] + xs[i] * Dp[d]) * gsil[i];
  yh[i] = (_Float16)v;
}

// ---------------------------------------------------------------------------
// WMMA f16 GEMM, 128x128x64 tiles, 8 waves (wave32), fused epilogues
//   mode 0: out0[row*N+col] = acc + bias[col]
//   mode 1: split projection: col<INTER -> xproj raw; else silu(gate)
//   mode 2: softplus + row-sum -> atomicAdd(rowsum[row])  (dt mean path)
//   mode 3: dout = resid + acc + bias   (final output)
// ---------------------------------------------------------------------------
__global__ __launch_bounds__(256) void k_gemm_wmma(const _Float16* __restrict__ A,
                                                   const _Float16* __restrict__ Bw,
                                                   const float* __restrict__ bias,
                                                   float* __restrict__ out0,
                                                   float* __restrict__ out1,
                                                   float* __restrict__ rowsum,
                                                   const float* __restrict__ resid,
                                                   float* __restrict__ dout,
                                                   int M, int N, int K, int mode) {
  __shared__ _Float16 As[BM * AK];
  __shared__ _Float16 Bs[BN * AK];     // stored transposed: [n][k]
  __shared__ float    sRow[BM];

  const int tid  = threadIdx.x;
  const int lane = tid & 31;
  const int wave = tid >> 5;
  const int wm   = wave >> 1;          // 0..3 : 32-row strip
  const int wn   = wave & 1;           // 0..1 : 64-col strip
  const int rowbase = blockIdx.y * BM;
  const int colbase = blockIdx.x * BN;

  if (mode == 2 && tid < BM) sRow[tid] = 0.0f;

  v8f acc[2][4];
  #pragma unroll
  for (int mt = 0; mt < 2; ++mt)
    #pragma unroll
    for (int nt = 0; nt < 4; ++nt)
      #pragma unroll
      for (int e = 0; e < 8; ++e) acc[mt][nt][e] = 0.0f;

  union V8 { uint4v u4; _Float16 h[8]; };

  const int  r16 = lane & 15;
  const bool hi  = lane >= 16;

  for (int kt = 0; kt < K; kt += BK) {
    // prefetch next A tile (global_prefetch_b8)
    if (kt + BK < K)
      __builtin_prefetch((const void*)(A + (size_t)(rowbase + (tid >> 1)) * K + kt + BK), 0, 1);

    // ---- stage A tile: BM x BK, row-major ----
#ifdef USE_ASYNC_COPY
    // gfx1250 async copy: global -> LDS directly, tracked by ASYNCcnt.
    // Rows are clamped (never out of range for our shapes; clamped rows feed
    // accumulator rows that the epilogue never stores).
    #pragma unroll
    for (int i = 0; i < 4; ++i) {
      int id = tid + i * 256;
      int r  = id >> 3;
      int c8 = (id & 7) * 8;
      int grow = rowbase + r;
      if (grow >= M) grow = M - 1;
      __builtin_amdgcn_global_load_async_to_lds_b128(
          (AS1 gint4*)(AS1 void*)(const void*)(A + (size_t)grow * K + kt + c8),
          (AS3 gint4*)(AS3 void*)(void*)(&As[r * AK + c8]), 0, 0);
    }
#else
    #pragma unroll
    for (int i = 0; i < 4; ++i) {
      int id = tid + i * 256;
      int r  = id >> 3;
      int c8 = (id & 7) * 8;
      V8 v;
      int grow = rowbase + r;
      if (grow < M) {
        v.u4 = *(const uint4v*)(A + (size_t)grow * K + kt + c8);
      } else {
        for (int j = 0; j < 8; ++j) v.h[j] = (_Float16)0.0f;
      }
      *(uint4v*)(&As[r * AK + c8]) = v.u4;
    }
#endif
    // ---- stage B tile transposed: Bs[n][k] from Bw[k][n] ----
    #pragma unroll
    for (int i = 0; i < 4; ++i) {
      int id = tid + i * 256;
      int kk = id >> 4;
      int n8 = (id & 15) * 8;
      V8 v;
      if (colbase + n8 + 8 <= N) {
        v.u4 = *(const uint4v*)(Bw + (size_t)(kt + kk) * N + colbase + n8);
      } else {
        for (int j = 0; j < 8; ++j) v.h[j] = (_Float16)0.0f;
      }
      #pragma unroll
      for (int j = 0; j < 8; ++j) Bs[(n8 + j) * AK + kk] = v.h[j];
    }
#ifdef USE_ASYNC_COPY
    __builtin_amdgcn_s_wait_asynccnt(0);
#endif
    __syncthreads();

    // ---- two 16x16x32 k-steps per staged tile ----
    #pragma unroll
    for (int ks = 0; ks < BK; ks += 32) {
      union FragU { v16h v; unsigned u[8]; };
      FragU afrag[2], bfrag[4];
      const int selw = hi ? 8 : 0;     // dword offset: halfs {0..15} vs {16..31}
      #pragma unroll
      for (int mt = 0; mt < 2; ++mt) {
        int arow = wm * 32 + mt * 16 + r16;
        const unsigned* p = (const unsigned*)(&As[arow * AK + ks]);
        #pragma unroll
        for (int i = 0; i < 4; ++i) {
          afrag[mt].u[i]     = p[selw + i];        // K = sel..sel+7
          afrag[mt].u[4 + i] = p[4 + selw + i];    // K = sel+8..sel+15
        }
      }
      #pragma unroll
      for (int nt = 0; nt < 4; ++nt) {
        int bcol = wn * 64 + nt * 16 + r16;
        const unsigned* p = (const unsigned*)(&Bs[bcol * AK + ks + (hi ? 16 : 0)]);
        #pragma unroll
        for (int i = 0; i < 8; ++i) bfrag[nt].u[i] = p[i];  // 16 contiguous K
      }
      #pragma unroll
      for (int mt = 0; mt < 2; ++mt)
        #pragma unroll
        for (int nt = 0; nt < 4; ++nt)
          acc[mt][nt] = __builtin_amdgcn_wmma_f32_16x16x32_f16(
              false, afrag[mt].v, false, bfrag[nt].v,
              (short)0, acc[mt][nt], false, false);
    }
    __syncthreads();
  }

  // ---- epilogue (C/D layout: lane = col within 16, VGPR e = row within 8) ----
  #pragma unroll
  for (int mt = 0; mt < 2; ++mt) {
    #pragma unroll
    for (int nt = 0; nt < 4; ++nt) {
      int col   = colbase + wn * 64 + nt * 16 + r16;
      int mrow0 = rowbase + wm * 32 + mt * 16 + (hi ? 8 : 0);
      #pragma unroll
      for (int e = 0; e < 8; ++e) {
        int row = mrow0 + e;
        if (row >= M || col >= N) continue;
        float v = acc[mt][nt][e];
        if (bias) v += bias[col];
        if (mode == 0) {
          out0[(size_t)row * N + col] = v;
        } else if (mode == 1) {
          if (col < INTER) out0[(size_t)row * INTER + col] = v;
          else             out1[(size_t)row * INTER + (col - INTER)] = silu_f(v);
        } else if (mode == 2) {
          atomicAdd(&sRow[row - rowbase], softplus_f(v));
        } else { // mode 3
          dout[(size_t)row * N + col] = resid[(size_t)row * N + col] + v;
        }
      }
    }
  }
  if (mode == 2) {
    __syncthreads();
    if (tid < BM) {
      int row = rowbase + tid;
      if (row < M) atomicAdd(&rowsum[row], sRow[tid]);
    }
  }
}

// ---------------------------------------------------------------------------
// Host-side orchestration
// ---------------------------------------------------------------------------
static inline size_t alignup(size_t v) { return (v + 255) & ~(size_t)255; }

extern "C" void kernel_launch(void* const* d_in, const int* in_sizes, int n_in,
                              void* d_out, int out_size, void* d_ws, size_t ws_size,
                              hipStream_t stream) {
  (void)in_sizes; (void)n_in; (void)out_size; (void)ws_size;
  const float* x         = (const float*)d_in[0];
  const float* condition = (const float*)d_in[1];
  const float* ln_g      = (const float*)d_in[2];
  const float* ln_b      = (const float*)d_in[3];
  const float* W_in      = (const float*)d_in[4];
  const float* b_in      = (const float*)d_in[5];
  const float* conv_w    = (const float*)d_in[6];
  const float* conv_b    = (const float*)d_in[7];
  const float* W_dt      = (const float*)d_in[8];
  const float* b_dt      = (const float*)d_in[9];
  const float* dt_bias   = (const float*)d_in[10];
  const float* A_log     = (const float*)d_in[11];
  const float* Dp        = (const float*)d_in[12];
  const float* W_B       = (const float*)d_in[13];
  const float* b_B       = (const float*)d_in[14];
  const float* W_C       = (const float*)d_in[15];
  const float* b_C       = (const float*)d_in[16];
  const float* W_cond    = (const float*)d_in[17];
  const float* b_cond    = (const float*)d_in[18];
  const float* W_out     = (const float*)d_in[19];
  const float* b_out     = (const float*)d_in[20];
  float* out = (float*)d_out;

  // ---- workspace carve ----
  char* base = (char*)d_ws;
  size_t off = 0;
  auto carve = [&](size_t bytes) { char* p = base + off; off += alignup(bytes); return p; };
  _Float16* W_in_h  = (_Float16*)carve((size_t)HID * N2 * 2);
  _Float16* W_dt_h  = (_Float16*)carve((size_t)INTER * INTER * 2);
  _Float16* W_out_h = (_Float16*)carve((size_t)INTER * HID * 2);
  _Float16* W_B_h   = (_Float16*)carve((size_t)INTER * STATE * 2);
  _Float16* W_C_h   = (_Float16*)carve((size_t)INTER * STATE * 2);
  _Float16* xn_h    = (_Float16*)carve((size_t)TOK * HID * 2);
  _Float16* xs_h    = (_Float16*)carve((size_t)TOK * INTER * 2);
  _Float16* yh      = (_Float16*)carve((size_t)TOK * INTER * 2);
  float* xp0     = (float*)carve((size_t)TOK * INTER * 4);
  float* xs      = (float*)carve((size_t)TOK * INTER * 4);
  float* gatesil = (float*)carve((size_t)TOK * INTER * 4);
  float* cond    = (float*)carve((size_t)BATCH * INTER * 4);
  float* dtb     = (float*)carve((size_t)INTER * 4);
  float* dtsum   = (float*)carve((size_t)TOK * 4);
  float* xmean   = (float*)carve((size_t)TOK * 4);
  float* Braw    = (float*)carve((size_t)TOK * STATE * 4);
  float* Cmat    = (float*)carve((size_t)TOK * STATE * 4);
  float* yscan   = (float*)carve((size_t)TOK * 4);

  auto cvt = [&](const float* s, _Float16* d, int n) {
    k_f32_to_f16<<<(n + 255) / 256, 256, 0, stream>>>(s, d, n);
  };

  // 1) precision staging + small precomputes
  cvt(W_in,  W_in_h,  HID * N2);
  cvt(W_dt,  W_dt_h,  INTER * INTER);
  cvt(W_out, W_out_h, INTER * HID);
  cvt(W_B,   W_B_h,   INTER * STATE);
  cvt(W_C,   W_C_h,   INTER * STATE);
  k_addvec<<<INTER / 256, 256, 0, stream>>>(b_dt, dt_bias, dtb, INTER);
  k_zero<<<TOK / 256, 256, 0, stream>>>(dtsum, TOK);
  k_cond_gemm<<<(BATCH * INTER) / 256, 256, 0, stream>>>(condition, W_cond, b_cond, cond);

  // 2) layernorm
  k_layernorm<<<TOK, 256, 0, stream>>>(x, ln_g, ln_b, xn_h);

  // 3) GEMM1: xn @ W_in -> xproj (raw) + silu(gate)
  k_gemm_wmma<<<dim3(N2 / BN, TOK / BM), 256, 0, stream>>>(
      xn_h, W_in_h, b_in, xp0, gatesil, nullptr, nullptr, nullptr,
      TOK, N2, HID, /*mode=*/1);

  // 4) depthwise conv3 + silu + cond + token mean
  k_conv_silu_cond<<<TOK, 256, 0, stream>>>(xp0, conv_w, conv_b, cond, xs, xs_h, xmean);

  // 5) dt path: softplus(xs @ W_dt + (b_dt + dt_bias)) row-summed into dtsum
  k_gemm_wmma<<<dim3(INTER / BN, TOK / BM), 256, 0, stream>>>(
      xs_h, W_dt_h, dtb, nullptr, nullptr, dtsum, nullptr, nullptr,
      TOK, INTER, INTER, /*mode=*/2);

  // 6) B and C projections (N=64 -> one col-block, guarded)
  k_gemm_wmma<<<dim3(1, TOK / BM), 256, 0, stream>>>(
      xs_h, W_B_h, b_B, Braw, nullptr, nullptr, nullptr, nullptr,
      TOK, STATE, INTER, /*mode=*/0);
  k_gemm_wmma<<<dim3(1, TOK / BM), 256, 0, stream>>>(
      xs_h, W_C_h, b_C, Cmat, nullptr, nullptr, nullptr, nullptr,
      TOK, STATE, INTER, /*mode=*/0);

  // 7) sequential scan
  k_scan<<<BATCH, STATE, 0, stream>>>(dtsum, xmean, Braw, Cmat, A_log, yscan);

  // 8) gated combine -> f16
  {
    size_t n = (size_t)TOK * INTER;
    k_build_y<<<(unsigned)((n + 255) / 256), 256, 0, stream>>>(yscan, xs, Dp, gatesil, yh);
  }

  // 9) output GEMM with residual add, straight into d_out
  k_gemm_wmma<<<dim3(HID / BN, TOK / BM), 256, 0, stream>>>(
      yh, W_out_h, b_out, nullptr, nullptr, nullptr, x, out,
      TOK, HID, INTER, /*mode=*/3);
}
